// HashedNetwork_89962384982886
// MI455X (gfx1250) — matrix operand order repaired and checked by reading, not compiled, not run on hardware
//
#include <hip/hip_runtime.h>

typedef __attribute__((ext_vector_type(16))) __bf16 bf16x16;
typedef __attribute__((ext_vector_type(8)))  __bf16 bf16x8;
typedef __attribute__((ext_vector_type(4)))  __bf16 bf16x4;
typedef __attribute__((ext_vector_type(8)))  float  floatx8;
typedef __attribute__((ext_vector_type(4)))  int    int4v;

typedef __attribute__((address_space(1))) int4v as1_int4;
typedef __attribute__((address_space(3))) int4v as3_int4;
typedef __attribute__((address_space(1))) void  as1_void;
typedef __attribute__((address_space(3))) void  as3_void;

#define BM 128
#define BN 128
#define BK 64     // two WMMA k-steps per LDS stage
#define LDST 72   // padded LDS row stride in bf16 elems (144B; keeps 16B chunks aligned)

union FragU { bf16x16 v; bf16x8 h[2]; };

__device__ __forceinline__ void wait_async_all() {
#if __has_builtin(__builtin_amdgcn_s_wait_asynccnt)
  __builtin_amdgcn_s_wait_asynccnt(0);
#else
  asm volatile("s_wait_asynccnt 0" ::: "memory");
#endif
}

// ---------------- fp32 -> bf16 conversion ----------------
__global__ __launch_bounds__(256) void f32_to_bf16_kernel(const float* __restrict__ in,
                                                          __bf16* __restrict__ out, int n4) {
  int i = blockIdx.x * blockDim.x + threadIdx.x;
  if (i < n4) {
    const float4 v = ((const float4*)in)[i];
    bf16x4 o = { (__bf16)v.x, (__bf16)v.y, (__bf16)v.z, (__bf16)v.w };
    ((bf16x4*)out)[i] = o;
  }
}

// ---------------- fused GEMM + bias (+ReLU) ----------------
// C[m,n] = sum_k A[m,k]*W[n,k] + bias[n]   (torch Linear: x @ W^T + b)
// A: [M,K] bf16 row-major, W: [N,K] bf16 row-major, C: [M,N] OutT
template <bool RELU, typename OutT>
__global__ __launch_bounds__(256) void gemm_bias_kernel(
    const __bf16* __restrict__ A,
    const __bf16* __restrict__ W,
    const float*  __restrict__ bias,
    OutT* __restrict__ C,
    int M, int N, int K)
{
  __shared__ alignas(16) __bf16 sA[2][BM * LDST];
  __shared__ alignas(16) __bf16 sB[2][BN * LDST];

  const int tid   = threadIdx.x;
  const int lane  = tid & 31;
  const int wid   = tid >> 5;
  const int waveM = wid >> 2;   // 0..1  (64 rows each)
  const int waveN = wid & 3;    // 0..3  (32 cols each)
  const int r     = lane & 15;
  const int hi    = lane >> 4;

  const int bn = blockIdx.x;
  const int bm = blockIdx.y;

  floatx8 acc[4][2];
  const floatx8 zero = {0.f,0.f,0.f,0.f,0.f,0.f,0.f,0.f};
  #pragma unroll
  for (int i = 0; i < 4; ++i)
    #pragma unroll
    for (int j = 0; j < 2; ++j)
      acc[i][j] = zero;

  // Stage copy: 128 rows x 64 cols bf16 per tile = 1024 b128 chunks; 256 threads
  // -> each thread moves 4 chunks per tile (rows r0+{0,32,64,96}, fixed 16B chunk)
  const int r0  = tid >> 3;        // 0..31
  const int cc8 = (tid & 7) * 8;   // bf16 element offset of this thread's chunk

  auto issue_stage = [&](int kt, int buf) {
    const int k0 = kt * BK;
    #pragma unroll
    for (int s = 0; s < 4; ++s) {
      const int row = r0 + s * 32;
      const __bf16* ga = A + (size_t)(bm * BM + row) * K + k0 + cc8;
      const __bf16* gw = W + (size_t)(bn * BN + row) * K + k0 + cc8;
      __bf16* la = &sA[buf][row * LDST + cc8];
      __bf16* lw = &sB[buf][row * LDST + cc8];
#if __has_builtin(__builtin_amdgcn_global_load_async_to_lds_b128)
      __builtin_amdgcn_global_load_async_to_lds_b128(
          (as1_int4*)(as1_void*)ga, (as3_int4*)(as3_void*)la, 0, 0);
      __builtin_amdgcn_global_load_async_to_lds_b128(
          (as1_int4*)(as1_void*)gw, (as3_int4*)(as3_void*)lw, 0, 0);
#else
      *(bf16x8*)la = *(const bf16x8*)ga;
      *(bf16x8*)lw = *(const bf16x8*)gw;
#endif
    }
  };

  const int KT = K / BK;
  issue_stage(0, 0);

  for (int kt = 0; kt < KT; ++kt) {
    const int buf = kt & 1;
    wait_async_all();      // this wave's async copies for stage kt have landed
    __syncthreads();       // ...and every other wave's too; prior reads of buf^1 done

    if (kt + 1 < KT) issue_stage(kt + 1, buf ^ 1);  // overlap next copies with math

    #pragma unroll
    for (int kk = 0; kk < 2; ++kk) {
      const int ko = kk * 32;
      // A fragment (16x32): lane r,hi -> K = hi*8..+7 in v[0:3], 16+hi*8..+7 in v[4:7]
      FragU fa[4];
      #pragma unroll
      for (int i = 0; i < 4; ++i) {
        const __bf16* p = &sA[buf][(waveM * 64 + i * 16 + r) * LDST + ko + hi * 8];
        fa[i].h[0] = *(const bf16x8*)p;
        fa[i].h[1] = *(const bf16x8*)(p + 16);
      }
      // B fragment (32x16): lane r = column n (= row of W), K = hi*8..+7 in v[0:3], 16+hi*8..+7 in v[4:7]
      FragU fb[2];
      #pragma unroll
      for (int j = 0; j < 2; ++j) {
        const __bf16* p = &sB[buf][(waveN * 32 + j * 16 + r) * LDST + ko + hi * 8];
        fb[j].h[0] = *(const bf16x8*)p;
        fb[j].h[1] = *(const bf16x8*)(p + 16);
      }
      #pragma unroll
      for (int i = 0; i < 4; ++i)
        #pragma unroll
        for (int j = 0; j < 2; ++j)
          acc[i][j] = __builtin_amdgcn_wmma_f32_16x16x32_bf16(
              false, fa[i].v, false, fb[j].v, (short)0, acc[i][j], false, false);
    }
    __syncthreads();  // all waves done reading buf before it is refilled
  }

  // Epilogue: C/D layout -> lanes 0-15: M=v, lanes 16-31: M=8+v, N=lane&15
  #pragma unroll
  for (int i = 0; i < 4; ++i) {
    const int mrow = bm * BM + waveM * 64 + i * 16 + hi * 8;
    #pragma unroll
    for (int j = 0; j < 2; ++j) {
      const int ncol = bn * BN + waveN * 32 + j * 16 + r;
      const float bv = bias[ncol];
      #pragma unroll
      for (int v = 0; v < 8; ++v) {
        float val = acc[i][j][v] + bv;
        if (RELU) val = fmaxf(val, 0.0f);
        C[(size_t)(mrow + v) * N + ncol] = (OutT)val;
      }
    }
  }
}

// ---------------- launcher ----------------
extern "C" void kernel_launch(void* const* d_in, const int* in_sizes, int n_in,
                              void* d_out, int out_size, void* d_ws, size_t ws_size,
                              hipStream_t stream) {
  constexpr int B = 4096, IN = 1024, H = 4096, OUT = 1024;
  const float* x  = (const float*)d_in[0];
  const float* W1 = (const float*)d_in[1];
  const float* b1 = (const float*)d_in[2];
  const float* W2 = (const float*)d_in[3];
  const float* b2 = (const float*)d_in[4];
  const float* W3 = (const float*)d_in[5];
  const float* b3 = (const float*)d_in[6];
  const float* W4 = (const float*)d_in[7];
  const float* b4 = (const float*)d_in[8];
  float* out = (float*)d_out;

  char* ws = (char*)d_ws;
  size_t off = 0;
  auto wsalloc = [&](size_t bytes) { char* p = ws + off; off += bytes; return p; };
  __bf16* xb  = (__bf16*)wsalloc((size_t)B * IN * 2);
  __bf16* w1b = (__bf16*)wsalloc((size_t)H * IN * 2);
  __bf16* w2b = (__bf16*)wsalloc((size_t)H * H * 2);
  __bf16* w3b = (__bf16*)wsalloc((size_t)H * H * 2);
  __bf16* w4b = (__bf16*)wsalloc((size_t)OUT * H * 2);
  __bf16* h1b = (__bf16*)wsalloc((size_t)B * H * 2);   // layers 1 & 3 output
  __bf16* h2b = (__bf16*)wsalloc((size_t)B * H * 2);   // layer 2 output

  auto conv = [&](const float* src, __bf16* dst, size_t n) {
    int n4 = (int)(n / 4);
    f32_to_bf16_kernel<<<(n4 + 255) / 256, 256, 0, stream>>>(src, dst, n4);
  };
  conv(x,  xb,  (size_t)B * IN);
  conv(W1, w1b, (size_t)H * IN);
  conv(W2, w2b, (size_t)H * H);
  conv(W3, w3b, (size_t)H * H);
  conv(W4, w4b, (size_t)OUT * H);

  dim3 blk(256);
  gemm_bias_kernel<true,  __bf16><<<dim3(H / BN,   B / BM), blk, 0, stream>>>(xb,  w1b, b1, h1b, B, H,   IN);
  gemm_bias_kernel<true,  __bf16><<<dim3(H / BN,   B / BM), blk, 0, stream>>>(h1b, w2b, b2, h2b, B, H,   H);
  gemm_bias_kernel<true,  __bf16><<<dim3(H / BN,   B / BM), blk, 0, stream>>>(h2b, w3b, b3, h1b, B, H,   H);
  gemm_bias_kernel<false, float ><<<dim3(OUT / BN, B / BM), blk, 0, stream>>>(h1b, w4b, b4, out, B, OUT, H);
}
